// MultiHeadAttention_30571577213803
// MI455X (gfx1250) — matrix operand order repaired and checked by reading, not compiled, not run on hardware
//
#include <hip/hip_runtime.h>

// ---------------------------------------------------------------------------
// MHA for MI455X (gfx1250): bf16 WMMA pipeline, L2-resident LDS-free GEMMs
// with register double-buffering, TDM ping-pong staged flash attention.
//   cvt            : x f32 -> bf16
//   transpose_cvt  : w_qkv, w_out f32 [K,N] -> bf16 [N,K]
//   qkv_gemm       : xb @ wqkvT -> Q,K [B,H,T,D], V [B,H,D,T]  (bf16)
//   attn           : flash attention, K/V tiles double-buffered via
//                    tensor_load_to_lds + s_wait_tensorcnt
//   out_gemm       : attn_out @ woutT + b_out -> out f32
// ---------------------------------------------------------------------------

typedef __attribute__((ext_vector_type(16))) __bf16 v16bf;
typedef __attribute__((ext_vector_type(8)))  __bf16 v8bf;
typedef __attribute__((ext_vector_type(8)))  float  v8f;
typedef __attribute__((ext_vector_type(4)))  float  v4f;
typedef __attribute__((ext_vector_type(4)))  unsigned int v4u;
typedef __attribute__((ext_vector_type(8)))  int    v8i;
typedef __attribute__((ext_vector_type(4)))  int    v4i;
typedef __bf16 bf16;

#define H_  25
#define T_  2048
#define C_  1600
#define BT_ 4096   // B*T

__device__ __forceinline__ v16bf cat16(v8bf lo, v8bf hi) {
  return __builtin_shufflevector(lo, hi, 0,1,2,3,4,5,6,7,8,9,10,11,12,13,14,15);
}

__device__ __forceinline__ v8f wmma_bf16(v16bf a, v16bf b, v8f c) {
  return __builtin_amdgcn_wmma_f32_16x16x32_bf16(false, a, false, b, (short)0, c,
                                                 false, false);
}

// Per-wave LDS store->load ordering without a workgroup barrier.
__device__ __forceinline__ void wave_lds_fence() {
  asm volatile("s_wait_dscnt 0" ::: "memory");
}

__device__ __forceinline__ float rmax16(float v) {
  v = fmaxf(v, __shfl_xor(v, 1));
  v = fmaxf(v, __shfl_xor(v, 2));
  v = fmaxf(v, __shfl_xor(v, 4));
  v = fmaxf(v, __shfl_xor(v, 8));
  return v;
}
__device__ __forceinline__ float rsum16(float v) {
  v += __shfl_xor(v, 1);
  v += __shfl_xor(v, 2);
  v += __shfl_xor(v, 4);
  v += __shfl_xor(v, 8);
  return v;
}

// ---------------------------------------------------------------------------
// Tensor Data Mover: 2D tile load (data_size = 2 bytes) via D# descriptor.
// ---------------------------------------------------------------------------
__device__ __forceinline__ void tdm_load_2d(unsigned lds_off, const void* gptr,
                                            unsigned tile_d0, unsigned tile_d1,
                                            unsigned tensor_d0, unsigned tensor_d1,
                                            unsigned long long stride0) {
  unsigned long long ga = (unsigned long long)(uintptr_t)gptr;
  v4u g0;
  g0[0] = 1u;                                            // count=1
  g0[1] = lds_off;                                       // lds_addr (bytes)
  g0[2] = (unsigned)(ga & 0xffffffffu);                  // global_addr[31:0]
  g0[3] = (unsigned)((ga >> 32) & 0x01ffffffu) | (2u << 30);  // [56:32] | type=2
  v8i g1;
  g1[0] = (int)(1u << 16);                               // data_size=1 (2 bytes)
  g1[1] = (int)((tensor_d0 & 0xffffu) << 16);            // tensor_dim0[15:0]
  g1[2] = (int)(((tensor_d0 >> 16) & 0xffffu) | ((tensor_d1 & 0xffffu) << 16));
  g1[3] = (int)(((tensor_d1 >> 16) & 0xffffu) | ((tile_d0 & 0xffffu) << 16));
  g1[4] = (int)(tile_d1 & 0xffffu);                      // tile_dim1, tile_dim2=0
  g1[5] = (int)(stride0 & 0xffffffffu);                  // dim0 stride [31:0]
  g1[6] = (int)((stride0 >> 32) & 0xffffu);              // dim0 stride [47:32]
  g1[7] = 0;
  v4i g2 = {0, 0, 0, 0};
  v4i g3 = {0, 0, 0, 0};
#if defined(__clang_major__) && (__clang_major__ >= 23)
  v8i gz = {0, 0, 0, 0, 0, 0, 0, 0};
  __builtin_amdgcn_tensor_load_to_lds(g0, g1, g2, g3, gz, 0);
#else
  __builtin_amdgcn_tensor_load_to_lds(g0, g1, g2, g3, 0);
#endif
}

// ---------------------------------------------------------------------------
// f32 -> bf16 convert (n multiple of 4)
// ---------------------------------------------------------------------------
__global__ __launch_bounds__(256) void cvt_kernel(const float* __restrict__ src,
                                                  bf16* __restrict__ dst, int n) {
  int i = (blockIdx.x * 256 + threadIdx.x) * 4;
  if (i < n) {
    v4f f = *(const v4f*)(src + i);
    dst[i + 0] = (bf16)f[0];
    dst[i + 1] = (bf16)f[1];
    dst[i + 2] = (bf16)f[2];
    dst[i + 3] = (bf16)f[3];
  }
}

// ---------------------------------------------------------------------------
// f32 [K x N] -> bf16 [N x K] transpose+convert.  K, N multiples of 32.
// block (32, 8), grid (N/32, K/32)
// ---------------------------------------------------------------------------
__global__ __launch_bounds__(256) void transpose_cvt_kernel(
    const float* __restrict__ src, bf16* __restrict__ dst, int K, int N) {
  __shared__ float tile[32][33];
  int n = blockIdx.x * 32 + threadIdx.x;
  int k = blockIdx.y * 32 + threadIdx.y;
#pragma unroll
  for (int j = 0; j < 32; j += 8)
    tile[threadIdx.y + j][threadIdx.x] = src[(size_t)(k + j) * N + n];
  __syncthreads();
  int k2 = blockIdx.y * 32 + threadIdx.x;
  int n2 = blockIdx.x * 32 + threadIdx.y;
#pragma unroll
  for (int j = 0; j < 32; j += 8)
    dst[(size_t)(n2 + j) * K + k2] = (bf16)tile[threadIdx.x][threadIdx.y + j];
}

// ---------------------------------------------------------------------------
// Shared GEMM helpers: wave tile 64x32 (4 M-frags x 2 N-frags).
// ---------------------------------------------------------------------------
__device__ __forceinline__ void load_frags(const bf16* __restrict__ pa0,
                                           const bf16* __restrict__ pb0,
                                           int k0, int kb, int ks, int Kdim,
                                           v16bf af[4], v16bf bf_[2]) {
#pragma unroll
  for (int m = 0; m < 4; ++m) {
    const bf16* pa = pa0 + (size_t)m * 16 * Kdim + k0;
    af[m] = cat16(*(const v8bf*)(pa + kb), *(const v8bf*)(pa + kb + 16));
  }
#pragma unroll
  for (int n = 0; n < 2; ++n) {
    const bf16* pb = pb0 + (size_t)n * 16 * Kdim + k0;
    bf_[n] = cat16(*(const v8bf*)(pb + kb), *(const v8bf*)(pb + kb + 16));
  }
}

__device__ __forceinline__ void wmma_all(const v16bf af[4], const v16bf bf_[2],
                                         v8f acc[4][2]) {
#pragma unroll
  for (int m = 0; m < 4; ++m)
#pragma unroll
    for (int n = 0; n < 2; ++n)
      acc[m][n] = wmma_bf16(af[m], bf_[n], acc[m][n]);
}

// ---------------------------------------------------------------------------
// QKV projection, LDS-free, register double-buffered (K step 64).
// grid = 64*150/8 = 1200 blocks of 256 threads (8 waves).
// ---------------------------------------------------------------------------
__global__ __launch_bounds__(256) void qkv_gemm_kernel(
    const bf16* __restrict__ Xb,   // [4096 x 1600]
    const bf16* __restrict__ WT,   // [4800 x 1600]  (w_qkv transposed)
    const float* __restrict__ bias,// [4800]
    bf16* __restrict__ Qb,         // [B*H, T, 64]
    bf16* __restrict__ Kb,         // [B*H, T, 64]
    bf16* __restrict__ Vb)         // [B*H, 64, T]
{
  const int Kdim = C_;
  const int lane = threadIdx.x & 31;
  const int gw   = blockIdx.x * 8 + (threadIdx.x >> 5);
  const int tm   = gw & 63;        // 64 M tiles of 64 rows
  const int tn   = gw >> 6;        // 150 N tiles of 32 cols
  const int l15  = lane & 15;
  const int kb   = (lane >> 4) << 3;   // A frag half: 0 / 8
  const int ks   = (lane >> 4) << 4;   // unused; B frags also use kb (0 / 8)
  const int lrow = (lane >> 4) << 3;

  const bf16* pa0 = Xb + (size_t)(tm * 64 + l15) * Kdim;
  const bf16* pb0 = WT + (size_t)(tn * 32 + l15) * Kdim;

  v8f acc[4][2] = {};
  v16bf afc[4], bfc[2], afn[4], bfn[2];
  load_frags(pa0, pb0, 0, kb, ks, Kdim, afc, bfc);

  for (int k0 = 0; k0 < Kdim; k0 += 64) {
    if (k0 + 128 < Kdim) {
      __builtin_prefetch(pa0 + k0 + 128, 0, 3);
      __builtin_prefetch(pb0 + k0 + 128, 0, 3);
    }
    load_frags(pa0, pb0, k0 + 32, kb, ks, Kdim, afn, bfn);   // in flight
    wmma_all(afc, bfc, acc);                                 // compute current
    if (k0 + 64 < Kdim)
      load_frags(pa0, pb0, k0 + 64, kb, ks, Kdim, afc, bfc); // in flight
    wmma_all(afn, bfn, acc);
  }

  // epilogue: +bias, scatter into Q/K (BHTD) and V (BHDT)
#pragma unroll
  for (int m = 0; m < 4; ++m)
#pragma unroll
    for (int n = 0; n < 2; ++n)
#pragma unroll
      for (int r = 0; r < 8; ++r) {
        int grow = tm * 64 + m * 16 + lrow + r;        // t index 0..4095
        int gcol = tn * 32 + n * 16 + l15;             // 0..4799
        float v  = acc[m][n][r] + bias[gcol];
        int bb = grow >> 11, tt = grow & (T_ - 1);
        int region = gcol / C_;
        int cc = gcol - region * C_;
        int h = cc >> 6, d = cc & 63;
        bf16 hv = (bf16)v;
        size_t bh = (size_t)(bb * H_ + h);
        if (region == 0)      Qb[(bh * T_ + tt) * 64 + d] = hv;
        else if (region == 1) Kb[(bh * T_ + tt) * 64 + d] = hv;
        else                  Vb[(bh * 64 + d) * T_ + tt] = hv;
      }
}

// ---------------------------------------------------------------------------
// Flash attention.  grid = B*H*(T/64) = 1600, block = 128 (4 waves).
// K / V^T tiles ping-pong staged into LDS by the Tensor Data Mover: wave 0
// issues block j+1 loads, then s_wait_tensorcnt 2 guarantees block j's pair
// landed (TDM ops from one wave complete in order).
// ---------------------------------------------------------------------------
__global__ __launch_bounds__(128) void attn_kernel(
    const bf16* __restrict__ Qb,   // [B*H, T, 64]
    const bf16* __restrict__ Kb,   // [B*H, T, 64]
    const bf16* __restrict__ Vt,   // [B*H, 64, T]
    bf16* __restrict__ Ao)         // [B*T, C]
{
  __shared__ alignas(16) bf16 lK[2][32 * 64];   // 32 keys x 64 d, ping-pong
  __shared__ alignas(16) bf16 lV[2][64 * 32];   // 64 d x 32 keys, ping-pong
  __shared__ alignas(16) bf16 lP[4][16][40];    // per-wave P staging

  const int tid  = threadIdx.x;
  const int lane = tid & 31;
  const int wave = tid >> 5;
  const int bh   = blockIdx.x >> 5;
  const int qblk = blockIdx.x & 31;
  const int q0   = qblk * 64 + wave * 16;
  const int bb   = bh / H_, h = bh - bb * H_;
  const int l15  = lane & 15;
  const int lrow = (lane >> 4) << 3;
  const int kb   = (lane >> 4) << 3;
  const int ks   = (lane >> 4) << 4;

  const bf16* Kp = Kb + (size_t)bh * T_ * 64;
  const bf16* Vp = Vt + (size_t)bh * 64 * T_;

  // Q A-fragments (d split 0..31 / 32..63), direct from global
  v16bf qa[2];
  {
    const bf16* pq = Qb + ((size_t)bh * T_ + q0 + l15) * 64;
#pragma unroll
    for (int dsx = 0; dsx < 2; ++dsx)
      qa[dsx] = cat16(*(const v8bf*)(pq + dsx * 32 + kb),
                      *(const v8bf*)(pq + dsx * 32 + kb + 16));
  }

  float mst[8], lst[8];
#pragma unroll
  for (int r = 0; r < 8; ++r) { mst[r] = -1e30f; lst[r] = 0.f; }
  v8f o[4] = {};

  const unsigned lK_off[2] = {(unsigned)(uintptr_t)(void*)&lK[0][0],
                              (unsigned)(uintptr_t)(void*)&lK[1][0]};
  const unsigned lV_off[2] = {(unsigned)(uintptr_t)(void*)&lV[0][0],
                              (unsigned)(uintptr_t)(void*)&lV[1][0]};
  const int nblk = qblk * 2 + 2;   // key blocks of 32, block-uniform

  if (wave == 0) {
    tdm_load_2d(lK_off[0], Kp, 64, 32, 64, (unsigned)T_, 64ull);
    tdm_load_2d(lV_off[0], Vp, 32, 64, (unsigned)T_, 64, (unsigned long long)T_);
  }

  for (int i = 0; i < nblk; ++i) {
    const int j0 = i * 32;
    const int cur = i & 1;
    if (wave == 0) {
      if (i + 1 < nblk) {
        const int nxt = cur ^ 1;
        tdm_load_2d(lK_off[nxt], Kp + (size_t)(j0 + 32) * 64,
                    64, 32, 64, (unsigned)T_, 64ull);
        tdm_load_2d(lV_off[nxt], Vp + (j0 + 32),
                    32, 64, (unsigned)T_, 64, (unsigned long long)T_);
        __builtin_amdgcn_s_wait_tensorcnt(2);   // current pair complete
      } else {
        __builtin_amdgcn_s_wait_tensorcnt(0);
      }
    }
    __syncthreads();

    if (j0 <= q0 + 15) {
      const bf16* kt = &lK[cur][0];
      const bf16* vt = &lV[cur][0];

      v16bf kf[2][2];
#pragma unroll
      for (int n = 0; n < 2; ++n) {
        const bf16* pk = kt + (n * 16 + l15) * 64;
#pragma unroll
        for (int dsx = 0; dsx < 2; ++dsx)
          kf[n][dsx] = cat16(*(const v8bf*)(pk + dsx * 32 + kb),
                             *(const v8bf*)(pk + dsx * 32 + kb + 16));
      }

      v8f s0 = {}, s1 = {};
      s0 = wmma_bf16(qa[0], kf[0][0], s0);
      s0 = wmma_bf16(qa[1], kf[0][1], s0);
      s1 = wmma_bf16(qa[0], kf[1][0], s1);
      s1 = wmma_bf16(qa[1], kf[1][1], s1);

      float p0[8], p1[8], alpha[8];
#pragma unroll
      for (int r = 0; r < 8; ++r) {
        int qi = q0 + lrow + r;
        int ki = j0 + l15;
        float v0 = s0[r] * 0.125f;   // 1/sqrt(64)
        float v1 = s1[r] * 0.125f;
        if (ki > qi)      v0 = -1e30f;
        if (ki + 16 > qi) v1 = -1e30f;
        float mx = rmax16(fmaxf(v0, v1));
        float mn = fmaxf(mst[r], mx);
        float al = __expf(mst[r] - mn);
        float e0 = __expf(v0 - mn);
        float e1 = __expf(v1 - mn);
        float sm = rsum16(e0 + e1);
        lst[r]   = lst[r] * al + sm;
        mst[r]   = mn;
        alpha[r] = al;
        p0[r] = e0;
        p1[r] = e1;
      }
#pragma unroll
      for (int t = 0; t < 4; ++t)
#pragma unroll
        for (int r = 0; r < 8; ++r)
          o[t][r] = o[t][r] * alpha[r];

      // C-layout -> A-layout relayout of P through wave-private LDS
#pragma unroll
      for (int r = 0; r < 8; ++r) {
        lP[wave][lrow + r][l15]      = (bf16)p0[r];
        lP[wave][lrow + r][16 + l15] = (bf16)p1[r];
      }
      wave_lds_fence();
      v16bf pa;
      {
        const bf16* pp = &lP[wave][l15][0];
        pa = cat16(*(const v8bf*)(pp + kb), *(const v8bf*)(pp + kb + 16));
      }
      wave_lds_fence();

#pragma unroll
      for (int t = 0; t < 4; ++t) {
        const bf16* pv = vt + (t * 16 + l15) * 32;
        v16bf vf = cat16(*(const v8bf*)(pv + kb), *(const v8bf*)(pv + kb + 16));
        o[t] = wmma_bf16(pa, vf, o[t]);
      }
    }
    __syncthreads();   // reads drained before this buffer is refilled
  }

  // normalize and write [B,T,C] bf16
#pragma unroll
  for (int r = 0; r < 8; ++r) {
    int qi = q0 + lrow + r;
    float inv = 1.0f / lst[r];
    size_t base = ((size_t)bb * T_ + qi) * C_ + h * 64;
#pragma unroll
    for (int t = 0; t < 4; ++t)
      Ao[base + t * 16 + l15] = (bf16)(o[t][r] * inv);
  }
}

// ---------------------------------------------------------------------------
// Output projection, LDS-free, register double-buffered.
// grid = 64*50/8 = 400 blocks of 256 threads.
// ---------------------------------------------------------------------------
__global__ __launch_bounds__(256) void out_gemm_kernel(
    const bf16* __restrict__ Ain,   // [4096 x 1600] bf16
    const bf16* __restrict__ WT,    // [1600 x 1600] (w_out transposed)
    const float* __restrict__ bias, // [1600]
    float* __restrict__ out)        // [4096 x 1600] f32
{
  const int Kdim = C_;
  const int lane = threadIdx.x & 31;
  const int gw   = blockIdx.x * 8 + (threadIdx.x >> 5);
  const int tm   = gw & 63;        // 64 M tiles
  const int tn   = gw >> 6;        // 50 N tiles
  const int l15  = lane & 15;
  const int kb   = (lane >> 4) << 3;
  const int ks   = (lane >> 4) << 4;
  const int lrow = (lane >> 4) << 3;

  const bf16* pa0 = Ain + (size_t)(tm * 64 + l15) * Kdim;
  const bf16* pb0 = WT + (size_t)(tn * 32 + l15) * Kdim;

  v8f acc[4][2] = {};
  v16bf afc[4], bfc[2], afn[4], bfn[2];
  load_frags(pa0, pb0, 0, kb, ks, Kdim, afc, bfc);

  for (int k0 = 0; k0 < Kdim; k0 += 64) {
    if (k0 + 128 < Kdim) {
      __builtin_prefetch(pa0 + k0 + 128, 0, 3);
      __builtin_prefetch(pb0 + k0 + 128, 0, 3);
    }
    load_frags(pa0, pb0, k0 + 32, kb, ks, Kdim, afn, bfn);
    wmma_all(afc, bfc, acc);
    if (k0 + 64 < Kdim)
      load_frags(pa0, pb0, k0 + 64, kb, ks, Kdim, afc, bfc);
    wmma_all(afn, bfn, acc);
  }

#pragma unroll
  for (int m = 0; m < 4; ++m)
#pragma unroll
    for (int n = 0; n < 2; ++n)
#pragma unroll
      for (int r = 0; r < 8; ++r) {
        int grow = tm * 64 + m * 16 + lrow + r;
        int gcol = tn * 32 + n * 16 + l15;
        out[(size_t)grow * Kdim + gcol] = acc[m][n][r] + bias[gcol];
      }
}

// ---------------------------------------------------------------------------
extern "C" void kernel_launch(void* const* d_in, const int* in_sizes, int n_in,
                              void* d_out, int out_size, void* d_ws, size_t ws_size,
                              hipStream_t stream) {
  const float* x     = (const float*)d_in[0];
  const float* w_qkv = (const float*)d_in[1];
  const float* b_qkv = (const float*)d_in[2];
  const float* w_out = (const float*)d_in[3];
  const float* b_out = (const float*)d_in[4];
  float* out = (float*)d_out;

  // workspace layout (bf16 elements)
  const size_t n_x   = (size_t)BT_ * C_;        // 6,553,600
  const size_t n_wq  = (size_t)C_ * 3 * C_;     // 7,680,000
  const size_t n_wo  = (size_t)C_ * C_;         // 2,560,000
  const size_t n_qkv = (size_t)2 * H_ * T_ * 64;// 6,553,600 each
  bf16* xb  = (bf16*)d_ws;
  bf16* wqT = xb + n_x;
  bf16* woT = wqT + n_wq;
  bf16* Qw  = woT + n_wo;
  bf16* Kw  = Qw + n_qkv;
  bf16* Vw  = Kw + n_qkv;
  bf16* Aw  = Vw + n_qkv;

  cvt_kernel<<<(int)(n_x / 4 / 256), 256, 0, stream>>>(x, xb, (int)n_x);
  transpose_cvt_kernel<<<dim3(150, 50), dim3(32, 8), 0, stream>>>(w_qkv, wqT, C_, 3 * C_);
  transpose_cvt_kernel<<<dim3(50, 50), dim3(32, 8), 0, stream>>>(w_out, woT, C_, C_);

  qkv_gemm_kernel<<<1200, 256, 0, stream>>>(xb, wqT, b_qkv, Qw, Kw, Vw);
  attn_kernel<<<1600, 128, 0, stream>>>(Qw, Kw, Vw, Aw);
  out_gemm_kernel<<<400, 256, 0, stream>>>(Aw, woT, b_out, out);
}
